// AgglutinativeAttention_86457691668608
// MI455X (gfx1250) — matrix operand order repaired and checked by reading, not compiled, not run on hardware
//
#include <hip/hip_runtime.h>
#include <hip/hip_bf16.h>
#include <math.h>

#define B_ 4
#define S_ 1024
#define H_ 1024
#define NH_ 16
#define HD_ 64
#define SCALE_ 0.125f          /* 1/sqrt(64) */
#define VERB_BIAS_ 2.0f
#define ROOT_BIAS_ 1.5f
#define SUFFIX_BIAS_ 1.2f

typedef __bf16 bf16_t;
typedef __attribute__((ext_vector_type(4)))  float  v4f;
typedef __attribute__((ext_vector_type(8)))  bf16_t v8bf;
typedef __attribute__((ext_vector_type(16))) bf16_t v16bf;
typedef __attribute__((ext_vector_type(8)))  float  v8f;

// A fragment (16x32 bf16, MxK): lane row = lane&15; lane<16 holds K {0..7,16..23},
// lane>=16 holds K {8..15,24..31}. rowp points at the row, k0 = (lane<16)?0:8.
__device__ __forceinline__ v16bf frag_a16(const bf16_t* rowp, int k0) {
  v8bf lo = *(const v8bf*)(rowp + k0);
  v8bf hi = *(const v8bf*)(rowp + k0 + 16);
  v16bf r;
#pragma unroll
  for (int i = 0; i < 8; ++i) { r[i] = lo[i]; r[8 + i] = hi[i]; }
  return r;
}

// B fragment (32x16 bf16, KxN): lane col = lane&15; lane<16 holds K {0..7,16..23},
// lane>=16 holds K {8..15,24..31}. colp points at the column's K-major
// data, k0 = (lane<16)?0:8.
__device__ __forceinline__ v16bf frag_b16(const bf16_t* colp, int k0) {
  v8bf lo = *(const v8bf*)(colp + k0);
  v8bf hi = *(const v8bf*)(colp + k0 + 16);
  v16bf r;
#pragma unroll
  for (int i = 0; i < 8; ++i) { r[i] = lo[i]; r[8 + i] = hi[i]; }
  return r;
}

__device__ __forceinline__ v8f wmma_bf16(v16bf a, v16bf b, v8f c) {
  return __builtin_amdgcn_wmma_f32_16x16x32_bf16(false, a, false, b, (short)0, c,
                                                 false, false);
}

// gfx1250 async global->LDS copy (ASYNCcnt-tracked). INST_OFFSET applies to
// both the global and LDS address, so one base covers offset:0 / offset:16.
__device__ __forceinline__ unsigned lds_off32(const void* p) {
  // generic LDS pointers carry the wave-relative LDS byte offset in addr[31:0]
  return (unsigned)(unsigned long long)p;
}
__device__ __forceinline__ void async_b128(unsigned lds_off,
                                           const void* gaddr) {
  asm volatile("global_load_async_to_lds_b128 %0, %1, off"
               :: "v"(lds_off), "v"((unsigned long long)gaddr)
               : "memory");
}
__device__ __forceinline__ void async_b128_off16(unsigned lds_off,
                                                 const void* gaddr) {
  asm volatile("global_load_async_to_lds_b128 %0, %1, off offset:16"
               :: "v"(lds_off), "v"((unsigned long long)gaddr)
               : "memory");
}
__device__ __forceinline__ void wait_async_le4() {
  asm volatile("s_wait_asynccnt 0x4" ::: "memory");
}

// ---------------------------------------------------------------------------
// Kernel 1: morpho bias precompute.
// ---------------------------------------------------------------------------
__global__ void bias_prep_kernel(const int* __restrict__ morpho,
                                 int* __restrict__ nearest,
                                 float* __restrict__ colbias) {
  const int b = blockIdx.x;
  for (int i = threadIdx.x; i < S_; i += blockDim.x) {
    const int mt = morpho[b * S_ + i];
    colbias[b * S_ + i] = (mt == 0 ? ROOT_BIAS_ * 0.5f : 0.0f) +
                          (mt == 1 ? SUFFIX_BIAS_ * 0.3f : 0.0f);
    int best = 1 << 30, idx = -1;
    for (int j = 0; j < S_; ++j) {
      if (morpho[b * S_ + j] == 2) {
        int d = (i > j) ? (i - j) : (j - i);
        if (d < best) { best = d; idx = j; }   // strict < keeps smallest j on tie
      }
    }
    nearest[b * S_ + i] = idx;
  }
}

// ---------------------------------------------------------------------------
// Kernel 2: tiled bf16 WMMA GEMM  Y[M,N] = X[M,K] @ W[K,N] + bias[N]
// Block tile 128x64, K-step 32, 8 waves; wave tile 32x32 -> 4 WMMAs/K-step.
// Double-buffered LDS: one barrier per K-step, next tile's global loads overlap
// current tile's WMMAs.
// ---------------------------------------------------------------------------
#define TM 128
#define TN 64
#define TK 32

template <bool XBF16>
__device__ __forceinline__ void gemm_stage(const void* __restrict__ Xin,
                                           const float* __restrict__ W,
                                           bf16_t* __restrict__ ldsA,
                                           bf16_t* __restrict__ ldsB,
                                           int mblock, int nblock, int kb,
                                           int K, int N, int tid) {
  const int arow = tid >> 1;            // 0..127
  const int akk = (tid & 1) * 16;       // 0 or 16
  const int bkk = tid >> 4;             // 0..15 (plus pass*16)
  const int bn0 = (tid & 15) * 4;       // 0..60
  if (XBF16) {
    const bf16_t* src =
        (const bf16_t*)Xin + (size_t)(mblock + arow) * K + kb + akk;
    v8bf x0 = *(const v8bf*)src;
    v8bf x1 = *(const v8bf*)(src + 8);
    *(v8bf*)&ldsA[arow * TK + akk] = x0;
    *(v8bf*)&ldsA[arow * TK + akk + 8] = x1;
  } else {
    const float* src =
        (const float*)Xin + (size_t)(mblock + arow) * K + kb + akk;
    v4f f0 = *(const v4f*)src;
    v4f f1 = *(const v4f*)(src + 4);
    v4f f2 = *(const v4f*)(src + 8);
    v4f f3 = *(const v4f*)(src + 12);
    v8bf x0, x1;
#pragma unroll
    for (int i = 0; i < 4; ++i) {
      x0[i] = (bf16_t)f0[i]; x0[4 + i] = (bf16_t)f1[i];
      x1[i] = (bf16_t)f2[i]; x1[4 + i] = (bf16_t)f3[i];
    }
    *(v8bf*)&ldsA[arow * TK + akk] = x0;
    *(v8bf*)&ldsA[arow * TK + akk + 8] = x1;
  }
#pragma unroll
  for (int p = 0; p < 2; ++p) {
    int kk = bkk + p * 16;
    v4f wv = *(const v4f*)(W + (size_t)(kb + kk) * N + nblock + bn0);
#pragma unroll
    for (int i = 0; i < 4; ++i) ldsB[(bn0 + i) * TK + kk] = (bf16_t)wv[i];
  }
}

template <int MODE, bool XBF16>
__global__ __launch_bounds__(256) void gemm_bf16_kernel(
    const void* __restrict__ Xin, const float* __restrict__ W,
    const float* __restrict__ bias, void* __restrict__ out, int M, int K,
    int N) {
  __shared__ __align__(16) bf16_t ldsA[2][TM * TK];  // [row][kk]
  __shared__ __align__(16) bf16_t ldsB[2][TN * TK];  // [col][kk]

  const int tid = threadIdx.x;
  const int mblock = blockIdx.y * TM;
  const int nblock = blockIdx.x * TN;
  const int w = tid >> 5;
  const int lane = tid & 31;
  const int hv = lane >> 4;
  const int ln = lane & 15;
  const int wm = (w >> 1) * 32;   // 0,32,64,96
  const int wn = (w & 1) * 32;    // 0,32
  const int ak0 = hv ? 8 : 0;
  const int bk0 = hv * 8;

  v8f c00 = {}, c01 = {}, c10 = {}, c11 = {};

  gemm_stage<XBF16>(Xin, W, ldsA[0], ldsB[0], mblock, nblock, 0, K, N, tid);
  int buf = 0;
  for (int kb = 0; kb < K; kb += TK) {
    __syncthreads();
    if (kb + TK < K)
      gemm_stage<XBF16>(Xin, W, ldsA[buf ^ 1], ldsB[buf ^ 1], mblock, nblock,
                        kb + TK, K, N, tid);
    v16bf a0 = frag_a16(&ldsA[buf][(wm + ln) * TK], ak0);
    v16bf a1 = frag_a16(&ldsA[buf][(wm + 16 + ln) * TK], ak0);
    v16bf b0 = frag_b16(&ldsB[buf][(wn + ln) * TK], bk0);
    v16bf b1 = frag_b16(&ldsB[buf][(wn + 16 + ln) * TK], bk0);
    c00 = wmma_bf16(a0, b0, c00);
    c01 = wmma_bf16(a0, b1, c01);
    c10 = wmma_bf16(a1, b0, c10);
    c11 = wmma_bf16(a1, b1, c11);
    buf ^= 1;
  }

  // ---- epilogue
  v8f cf[2][2] = {{c00, c01}, {c10, c11}};
#pragma unroll
  for (int mi = 0; mi < 2; ++mi)
#pragma unroll
    for (int ni = 0; ni < 2; ++ni)
#pragma unroll
      for (int r = 0; r < 8; ++r) {
        int grow = mblock + wm + mi * 16 + r + 8 * hv;
        int gcol = nblock + wn + ni * 16 + ln;
        float val = cf[mi][ni][r] + bias[gcol];
        if (MODE == 2) {
          ((float*)out)[(size_t)grow * N + gcol] = val;
        } else {
          int bb = grow / S_, s = grow % S_;
          int hh = gcol / HD_, d = gcol % HD_;
          size_t idx = (MODE == 0)
                           ? ((((size_t)bb * NH_ + hh) * S_ + s) * HD_ + d)
                           : ((((size_t)bb * NH_ + hh) * HD_ + d) * S_ + s);
          ((bf16_t*)out)[idx] = (bf16_t)val;
        }
      }
}

// ---------------------------------------------------------------------------
// Kernel 3: flash attention. 128 threads = 4 waves; each wave owns a 16-query
// tile; the block cooperatively stages 32-key K/V tiles via ASYNC global->LDS
// copies, double-buffered so HBM latency hides under WMMA+softmax.
// 8 WMMAs per 32 keys per wave.
// ---------------------------------------------------------------------------
__global__ __launch_bounds__(128) void attn_kernel(
    const bf16_t* __restrict__ q, const bf16_t* __restrict__ k,
    const bf16_t* __restrict__ vT, const int* __restrict__ nearest,
    const float* __restrict__ colbias, bf16_t* __restrict__ attn_out) {
  const int bh = blockIdx.x;
  const int b = bh / NH_, h = bh % NH_;
  const int tid = threadIdx.x;
  const int wave = tid >> 5;
  const int lane = tid & 31;
  const int hv = lane >> 4;
  const int ln = lane & 15;
  const int ak0 = hv ? 8 : 0;
  const int bk0 = hv * 8;
  const int qbase = blockIdx.y * 64 + wave * 16;

  __shared__ __align__(16) bf16_t ldsK[2][32 * HD_];   // [key][dim]
  __shared__ __align__(16) bf16_t ldsV[2][HD_ * 32];   // [dim][key]
  __shared__ __align__(16) bf16_t ldsP[4][16 * 32];    // per-wave P transpose

  const size_t headQK = ((size_t)b * NH_ + h) * S_;
  const size_t headV = ((size_t)b * NH_ + h) * HD_;

  // Q A-fragments (K = head dim, chunks 0..31 / 32..63).
  const bf16_t* qrow = q + (headQK + qbase + ln) * HD_;
  v16bf aq0 = frag_a16(qrow, ak0);
  v16bf aq1 = frag_a16(qrow + 32, ak0);

  int nrst[8];
#pragma unroll
  for (int r = 0; r < 8; ++r)
    nrst[r] = nearest[(size_t)b * S_ + qbase + r + 8 * hv];

  float mstate[8], lstate[8];
#pragma unroll
  for (int r = 0; r < 8; ++r) { mstate[r] = -INFINITY; lstate[r] = 0.0f; }
  v8f o[4] = {v8f{}, v8f{}, v8f{}, v8f{}};

  // Staging coordinates: 32 bytes (2 x b128) per thread per tile, K and V.
  const int skey = tid >> 2, sd0 = (tid & 3) * 16;  // K tile: 32 keys x 64 dims
  const int sdim = tid >> 1, sk0 = (tid & 1) * 16;  // V tile: 64 dims x 32 keys
  unsigned ldsKoff[2], ldsVoff[2];
#pragma unroll
  for (int i = 0; i < 2; ++i) {
    ldsKoff[i] = lds_off32(&ldsK[i][skey * HD_ + sd0]);
    ldsVoff[i] = lds_off32(&ldsV[i][sdim * 32 + sk0]);
  }

  // Prologue: issue tile 0 into buffer 0 (4 async b128 per thread).
  async_b128(ldsKoff[0], k + (headQK + 0 + skey) * HD_ + sd0);
  async_b128_off16(ldsKoff[0], k + (headQK + 0 + skey) * HD_ + sd0);
  async_b128(ldsVoff[0], vT + (headV + sdim) * S_ + 0 + sk0);
  async_b128_off16(ldsVoff[0], vT + (headV + sdim) * S_ + 0 + sk0);

  int buf = 0;
  for (int kb = 0; kb < S_; kb += 32) {
    // Issue next tile (wraps at the end; harmless deterministic prefetch).
    const int kbn = (kb + 32) & (S_ - 1);
    async_b128(ldsKoff[buf ^ 1], k + (headQK + kbn + skey) * HD_ + sd0);
    async_b128_off16(ldsKoff[buf ^ 1], k + (headQK + kbn + skey) * HD_ + sd0);
    async_b128(ldsVoff[buf ^ 1], vT + (headV + sdim) * S_ + kbn + sk0);
    async_b128_off16(ldsVoff[buf ^ 1], vT + (headV + sdim) * S_ + kbn + sk0);
    wait_async_le4();   // newest 4 may be in flight => current tile landed
    __syncthreads();

    const bf16_t* lK = ldsK[buf];
    const bf16_t* lV = ldsV[buf];

    // ---- scores: two 16-key N-tiles, K=64 head dim -> 2 WMMAs each
    v8f s0 = {}, s1 = {};
    s0 = wmma_bf16(aq0, frag_b16(&lK[ln * HD_], bk0), s0);
    s0 = wmma_bf16(aq1, frag_b16(&lK[ln * HD_ + 32], bk0), s0);
    s1 = wmma_bf16(aq0, frag_b16(&lK[(16 + ln) * HD_], bk0), s1);
    s1 = wmma_bf16(aq1, frag_b16(&lK[(16 + ln) * HD_ + 32], bk0), s1);

    // ---- scale + morpho bias
    const int col0 = kb + ln, col1 = kb + 16 + ln;
    const float cb0 = colbias[(size_t)b * S_ + col0];
    const float cb1 = colbias[(size_t)b * S_ + col1];
    float p0[8], p1[8], rmax[8];
#pragma unroll
    for (int r = 0; r < 8; ++r) {
      float x0 = s0[r] * SCALE_ + cb0 + ((nrst[r] == col0) ? VERB_BIAS_ : 0.0f);
      float x1 = s1[r] * SCALE_ + cb1 + ((nrst[r] == col1) ? VERB_BIAS_ : 0.0f);
      p0[r] = x0; p1[r] = x1; rmax[r] = fmaxf(x0, x1);
    }
#pragma unroll
    for (int m = 1; m < 16; m <<= 1)
#pragma unroll
      for (int r = 0; r < 8; ++r)
        rmax[r] = fmaxf(rmax[r], __shfl_xor(rmax[r], m, 32));

    float alpha[8];
#pragma unroll
    for (int r = 0; r < 8; ++r) {
      float mn = fmaxf(mstate[r], rmax[r]);
      alpha[r] = __expf(mstate[r] - mn);
      mstate[r] = mn;
      p0[r] = __expf(p0[r] - mn);
      p1[r] = __expf(p1[r] - mn);
    }
    float rs[8];
#pragma unroll
    for (int r = 0; r < 8; ++r) rs[r] = p0[r] + p1[r];
#pragma unroll
    for (int m = 1; m < 16; m <<= 1)
#pragma unroll
      for (int r = 0; r < 8; ++r) rs[r] += __shfl_xor(rs[r], m, 32);
#pragma unroll
    for (int r = 0; r < 8; ++r) lstate[r] = lstate[r] * alpha[r] + rs[r];
#pragma unroll
    for (int t = 0; t < 4; ++t)
#pragma unroll
      for (int r = 0; r < 8; ++r) o[t][r] *= alpha[r];

    // ---- transpose P (C layout) -> A layout via wave-private LDS region.
    // Wave-internal LDS ops are in-order; no block barrier needed here.
#pragma unroll
    for (int r = 0; r < 8; ++r) {
      ldsP[wave][(r + 8 * hv) * 32 + ln] = (bf16_t)p0[r];
      ldsP[wave][(r + 8 * hv) * 32 + 16 + ln] = (bf16_t)p1[r];
    }
    v16bf aP = frag_a16(&ldsP[wave][ln * 32], ak0);

    // ---- O += P @ V : K=32 keys, 4 N-tiles of 16 dims
#pragma unroll
    for (int t = 0; t < 4; ++t)
      o[t] = wmma_bf16(aP, frag_b16(&lV[(t * 16 + ln) * 32], bk0), o[t]);
    __syncthreads();   // everyone done reading buf before it is overwritten
    buf ^= 1;
  }

  // ---- normalize and write [b, s, h, d] (flattens to [B,S,H]) as bf16
  float inv[8];
#pragma unroll
  for (int r = 0; r < 8; ++r) inv[r] = 1.0f / lstate[r];
#pragma unroll
  for (int t = 0; t < 4; ++t)
#pragma unroll
    for (int r = 0; r < 8; ++r) {
      int row = qbase + r + 8 * hv;
      int dim = t * 16 + ln;
      attn_out[(((size_t)b * S_ + row) * NH_ + h) * HD_ + dim] =
          (bf16_t)(o[t][r] * inv[r]);
    }
}

// ---------------------------------------------------------------------------
extern "C" void kernel_launch(void* const* d_in, const int* in_sizes, int n_in,
                              void* d_out, int out_size, void* d_ws,
                              size_t ws_size, hipStream_t stream) {
  (void)in_sizes; (void)n_in; (void)out_size; (void)ws_size;
  const float* hs     = (const float*)d_in[0];
  const int*   morpho = (const int*)d_in[1];
  const float* Wq = (const float*)d_in[2];
  const float* bq = (const float*)d_in[3];
  const float* Wk = (const float*)d_in[4];
  const float* bk = (const float*)d_in[5];
  const float* Wv = (const float*)d_in[6];
  const float* bv = (const float*)d_in[7];
  const float* Wo = (const float*)d_in[8];
  const float* bo = (const float*)d_in[9];

  const size_t nqkv = (size_t)B_ * NH_ * S_ * HD_;  // 4,194,304 elements
  bf16_t* qbuf  = (bf16_t*)d_ws;
  bf16_t* kbuf  = qbuf + nqkv;
  bf16_t* vTbuf = kbuf + nqkv;
  bf16_t* aobuf = vTbuf + nqkv;
  int*    nearp = (int*)(aobuf + nqkv);
  float*  cbias = (float*)(nearp + B_ * S_);

  bias_prep_kernel<<<dim3(B_), dim3(256), 0, stream>>>(morpho, nearp, cbias);

  const int M = B_ * S_;
  dim3 gg(H_ / TN, M / TM);
  gemm_bf16_kernel<0, false><<<gg, 256, 0, stream>>>(hs, Wq, bq, qbuf, M, H_, H_);
  gemm_bf16_kernel<0, false><<<gg, 256, 0, stream>>>(hs, Wk, bk, kbuf, M, H_, H_);
  gemm_bf16_kernel<1, false><<<gg, 256, 0, stream>>>(hs, Wv, bv, vTbuf, M, H_, H_);

  attn_kernel<<<dim3(B_ * NH_, S_ / 64), dim3(128), 0, stream>>>(
      qbuf, kbuf, vTbuf, nearp, cbias, aobuf);

  gemm_bf16_kernel<2, true><<<gg, 256, 0, stream>>>(aobuf, Wo, bo, d_out, M, H_,
                                                    H_);
}
